// FiLM_74990128988772
// MI455X (gfx1250) — matrix operand order repaired and checked by reading, not compiled, not run on hardware
//
#include <hip/hip_runtime.h>
#include <hip/hip_bf16.h>
#include <math.h>

// ---------------------------------------------------------------------------
// FiLM classifier head for MI455X (gfx1250, wave32, WMMA).
// Scan is re-expressed as  xf[r,m,:] = sum_s f[r,s] * T1[s,m,:]  so the bulk
// of the FLOPs runs through v_wmma_f32_16x16x32_f16 GEMMs (f32 accumulate).
// Round 2: 128x64 GEMM blocks (4 WMMA per A-fragment per K-chunk) and
// global_prefetch of the next K-chunk while WMMAs run.
// ---------------------------------------------------------------------------

typedef __attribute__((ext_vector_type(16))) _Float16 v16h;
typedef __attribute__((ext_vector_type(8)))  float    v8f;

#define BATCHN 32
#define LSEQ   512
#define CIN    32
#define NH     128
#define MODES  32
#define KS     3
#define RROWS  (BATCHN*CIN)   // 1024
#define MN     (MODES*NH)     // 4096
#define KN     (KS*NH)        // 384
#define NCLS   10
#define FINF   (LSEQ*CIN)     // 16384

__device__ inline v8f zero8() {
  v8f v;
#pragma unroll
  for (int i = 0; i < 8; ++i) v[i] = 0.0f;
  return v;
}

__device__ inline v8f wmma_f16(v16h a, v16h b, v8f c) {
  // D = A(16x32,f16) x B(32x16,f16) + C(16x16,f32)
  return __builtin_amdgcn_wmma_f32_16x16x32_f16(false, a, false, b, (short)0, c,
                                                false, false);
}

// A fragment (16x32 f16) from row-major LDS tile. Contiguous k order (same as load_frag_b):
// lane: M = lane%16, half h = lane/16; element e -> K = e + 16*h
__device__ inline v16h load_frag_a(const _Float16* smem, int lda) {
  int lane = threadIdx.x & 31;
  int row = lane & 15;
  int h   = lane >> 4;
  v16h out;
#pragma unroll
  for (int e = 0; e < 16; ++e) {
    int k = e + 16 * h;
    out[e] = smem[row * lda + k];
  }
  return out;
}

// B fragment (32x16 f16) from an N-major (transposed) LDS tile Bs[n][k].
// lane: N = lane%16, half h = lane/16; element e -> K = e + 16*h
__device__ inline v16h load_frag_b(const _Float16* smem, int ldb) {
  int lane = threadIdx.x & 31;
  int n = lane & 15;
  int h = lane >> 4;
  v16h out;
#pragma unroll
  for (int e = 0; e < 16; ++e) {
    int k = e + 16 * h;
    out[e] = smem[n * ldb + k];
  }
  return out;
}

// ---------------------------------------------------------------------------
// 0) instance norm: mean/std over seq, affine, write f16 rows xn[r][s]
// ---------------------------------------------------------------------------
__global__ __launch_bounds__(256) void norm_kernel(
    const float* __restrict__ xe, const float* __restrict__ aw,
    const float* __restrict__ ab, _Float16* __restrict__ xnh,
    float* __restrict__ meanb, float* __restrict__ stdb) {
  int r = blockIdx.x;
  int b = r / CIN, c = r % CIN;
  __shared__ float red[256];
  float s = 0.f, s2 = 0.f;
  for (int t = threadIdx.x; t < LSEQ; t += 256) {
    float v = xe[(size_t)(b * LSEQ + t) * CIN + c];
    s += v; s2 += v * v;
  }
  red[threadIdx.x] = s; __syncthreads();
  for (int o = 128; o > 0; o >>= 1) {
    if (threadIdx.x < o) red[threadIdx.x] += red[threadIdx.x + o];
    __syncthreads();
  }
  float mean = red[0] / (float)LSEQ;
  __syncthreads();
  red[threadIdx.x] = s2; __syncthreads();
  for (int o = 128; o > 0; o >>= 1) {
    if (threadIdx.x < o) red[threadIdx.x] += red[threadIdx.x + o];
    __syncthreads();
  }
  float var = red[0] / (float)LSEQ - mean * mean;
  float sd = sqrtf(var + 1e-5f);
  if (threadIdx.x == 0) { meanb[r] = mean; stdb[r] = sd; }
  float w = aw[c], bb = ab[c];
  for (int t = threadIdx.x; t < LSEQ; t += 256) {
    float v = xe[(size_t)(b * LSEQ + t) * CIN + c];
    xnh[(size_t)r * LSEQ + t] = (_Float16)(((v - mean) / sd) * w + bb);
  }
}

// ---------------------------------------------------------------------------
// 1) G[d] = A^d B  (the only sequential piece; one block per scale k)
// ---------------------------------------------------------------------------
__global__ __launch_bounds__(512) void hippo_g_kernel(
    const float* __restrict__ A_stack, const float* __restrict__ B_stack,
    float* __restrict__ Gbuf) {
  int k = blockIdx.x;
  __shared__ float gcur[NH], gnext[NH], part[512];
  const float* Ak = A_stack + (size_t)k * NH * NH;
  if (threadIdx.x < NH) gcur[threadIdx.x] = B_stack[k * NH + threadIdx.x];
  __syncthreads();
  int n = threadIdx.x >> 2, sub = threadIdx.x & 3;
  float* Gk = Gbuf + (size_t)k * LSEQ * NH;
  for (int d = 0; d < LSEQ; ++d) {
    if (threadIdx.x < NH) Gk[(size_t)d * NH + threadIdx.x] = gcur[threadIdx.x];
    float p = 0.f;
    const float* arow = Ak + (size_t)n * NH + sub * 32;
    const float* gv = &gcur[sub * 32];
#pragma unroll 8
    for (int j = 0; j < 32; ++j) p += arow[j] * gv[j];
    part[threadIdx.x] = p;
    __syncthreads();
    if (sub == 0)
      gnext[n] = part[n * 4] + part[n * 4 + 1] + part[n * 4 + 2] + part[n * 4 + 3];
    __syncthreads();
    if (threadIdx.x < NH) gcur[threadIdx.x] = gnext[threadIdx.x];
    __syncthreads();
  }
}

// ---------------------------------------------------------------------------
// 2) T1[s,m,n] = W(m,s) * sum_{d<=L-1-s} W(m,d) G[d][n]   (W = e^{-2pi i m d/L})
// ---------------------------------------------------------------------------
__global__ __launch_bounds__(256) void t1_kernel(
    const float* __restrict__ Gbuf, _Float16* __restrict__ T1r,
    _Float16* __restrict__ T1i) {
  int k = blockIdx.y;
  int idx = blockIdx.x * 256 + threadIdx.x;  // 0..4095
  int m = idx >> 7, n = idx & 127;
  const float* Gk = Gbuf + (size_t)k * LSEQ * NH;
  _Float16* t1r = T1r + (size_t)k * LSEQ * MN;
  _Float16* t1i = T1i + (size_t)k * LSEQ * MN;
  float phr = 0.f, phi = 0.f;
  float w0 = -6.283185307179586f * (float)m / (float)LSEQ;
  for (int r = 0; r < LSEQ; ++r) {
    float g = Gk[(size_t)r * NH + n];
    float sr, cr;
    __sincosf(w0 * (float)r, &sr, &cr);
    phr += cr * g; phi += sr * g;
    int s = LSEQ - 1 - r;
    float ss, cs;
    __sincosf(w0 * (float)s, &ss, &cs);
    t1r[(size_t)s * MN + m * NH + n] = (_Float16)(cs * phr - ss * phi);
    t1i[(size_t)s * MN + m * NH + n] = (_Float16)(cs * phi + ss * phr);
  }
}

// ---------------------------------------------------------------------------
// 3) layout converters (f32 -> f16, GEMM-friendly layouts)
// ---------------------------------------------------------------------------
__global__ void wconv_kernel(const float* __restrict__ wr,
                             const float* __restrict__ wi,
                             _Float16* __restrict__ Whr,
                             _Float16* __restrict__ Whi) {
  size_t idx = (size_t)blockIdx.x * 256 + threadIdx.x;
  size_t total = (size_t)KS * MODES * NH * NH;
  if (idx >= total) return;
  int o = idx % NH; size_t t2 = idx / NH;
  int i = t2 % NH; size_t t3 = t2 / NH;
  int m = t3 % MODES; int k = (int)(t3 / MODES);
  size_t src = (((size_t)k * NH + i) * NH + o) * MODES + m;
  Whr[idx] = (_Float16)wr[src];
  Whi[idx] = (_Float16)wi[src];
}

__global__ void ew_kernel(const float* __restrict__ eval_last,
                          const float* __restrict__ mlp_w,
                          _Float16* __restrict__ EWh) {
  int idx = blockIdx.x * 256 + threadIdx.x;  // KN*LSEQ
  if (idx >= KN * LSEQ) return;
  int s = idx % LSEQ; int kn = idx / LSEQ;
  int k = kn / NH, n = kn % NH;
  EWh[idx] = (_Float16)(mlp_w[k] * eval_last[((size_t)k * LSEQ + s) * NH + n]);
}

__global__ void dec2h_kernel(const float* __restrict__ dec,
                             _Float16* __restrict__ DECh) {
  int idx = blockIdx.x * 256 + threadIdx.x;  // RROWS*KN
  if (idx >= RROWS * KN) return;
  int kn = idx % KN; int r = idx / KN;
  int k = kn / NH, n = kn % NH;
  DECh[idx] = (_Float16)dec[((size_t)k * RROWS + r) * NH + n];
}

__global__ void zero_kernel(float* __restrict__ p, int n) {
  int i = blockIdx.x * 256 + threadIdx.x;
  if (i < n) p[i] = 0.f;
}

// ---------------------------------------------------------------------------
// 4) DFT GEMM:  XF{r,i}[k] (1024 x 4096) = xn (1024 x 512) @ T1{r,i}[k]
//    128x64 block, BK=32, 8 waves; each wave: 1 A-frag -> 4 WMMA per chunk.
//    Prefetch of next K-chunk overlaps the matrix ops (L2-resident data).
// ---------------------------------------------------------------------------
__global__ __launch_bounds__(256) void gemm_dft_kernel(
    const _Float16* __restrict__ A, const _Float16* __restrict__ T1r,
    const _Float16* __restrict__ T1i, _Float16* __restrict__ XFr,
    _Float16* __restrict__ XFi) {
  int z = blockIdx.z;  // 0..5 -> (k, re/im)
  int k = z >> 1, part = z & 1;
  const _Float16* Bm = (part ? T1i : T1r) + (size_t)k * LSEQ * MN;
  _Float16* Cm = (part ? XFi : XFr) + (size_t)k * RROWS * MN;

  int colBase = blockIdx.x * 64;
  int rowBase = blockIdx.y * 128;

  __shared__ _Float16 As[128][40];
  __shared__ _Float16 Bs[64][40];  // Bs[n][k] (transposed)

  int t = threadIdx.x;
  int wave = t >> 5, lane = t & 31;

  v8f acc[4];
#pragma unroll
  for (int q = 0; q < 4; ++q) acc[q] = zero8();

  int rowA = t >> 1, cA = (t & 1) * 16;
  int nB = t & 63, krB = t >> 6;  // krB 0..3

  for (int kk = 0; kk < LSEQ; kk += 32) {
    const _Float16* srcA = A + (size_t)(rowBase + rowA) * LSEQ + kk + cA;
#pragma unroll
    for (int q = 0; q < 16; ++q) As[rowA][cA + q] = srcA[q];
#pragma unroll
    for (int q = 0; q < 8; ++q) {
      int kv = krB * 8 + q;
      Bs[nB][kv] = Bm[(size_t)(kk + kv) * MN + colBase + nB];
    }
    if (kk + 32 < LSEQ) {
      __builtin_prefetch(srcA + 32, 0, 0);
      __builtin_prefetch(&Bm[(size_t)(kk + 32 + krB * 8) * MN + colBase + nB], 0, 0);
    }
    __syncthreads();
    v16h aF = load_frag_a(&As[wave * 16][0], 40);
#pragma unroll
    for (int nt = 0; nt < 4; ++nt) {
      v16h bF = load_frag_b(&Bs[nt * 16][0], 40);
      acc[nt] = wmma_f16(aF, bF, acc[nt]);
    }
    __syncthreads();
  }
  int h = lane >> 4, nn = lane & 15;
#pragma unroll
  for (int nt = 0; nt < 4; ++nt) {
#pragma unroll
    for (int i = 0; i < 8; ++i) {
      int row = rowBase + wave * 16 + i + 8 * h;
      Cm[(size_t)row * MN + colBase + nt * 16 + nn] = (_Float16)acc[nt][i];
    }
  }
}

// ---------------------------------------------------------------------------
// 5) spectral GEMM + fused single-point irFFT:
//    per (k, m): OUT = (XFr + iXFi)[:, m*128:+128] @ (Wr + iWi)[m]
//    dec[k][r,o] += w_m * Re(OUT * e^{i*2pi*m*(L-1)/L})   (atomic over m-groups)
// ---------------------------------------------------------------------------
__global__ __launch_bounds__(256) void gemm_spec_kernel(
    const _Float16* __restrict__ XFr, const _Float16* __restrict__ XFi,
    const _Float16* __restrict__ Whr, const _Float16* __restrict__ Whi,
    float* __restrict__ dec) {
  int k = blockIdx.z;
  int rt = blockIdx.x >> 1, ot = blockIdx.x & 1;
  int rowBase = rt * 64, oBase = ot * 64;
  const _Float16* Arm = XFr + (size_t)k * RROWS * MN;
  const _Float16* Aim = XFi + (size_t)k * RROWS * MN;
  float* deck = dec + (size_t)k * RROWS * NH;

  __shared__ _Float16 Asr[64][40], Asi[64][40], Bsr[64][40], Bsi[64][40];

  int t = threadIdx.x, wave = t >> 5, lane = t & 31;
  int wm = wave >> 1, wn = wave & 1;
  int h = lane >> 4, nn = lane & 15;

  for (int mm = 0; mm < 4; ++mm) {
    int m = blockIdx.y * 4 + mm;
    const _Float16* Br = Whr + ((size_t)k * MODES + m) * NH * NH;
    const _Float16* Bi = Whi + ((size_t)k * MODES + m) * NH * NH;
    v8f aRR[2], aII[2], aRI[2], aIR[2];
#pragma unroll
    for (int q = 0; q < 2; ++q) {
      aRR[q] = zero8(); aII[q] = zero8(); aRI[q] = zero8(); aIR[q] = zero8();
    }
    for (int kk = 0; kk < NH; kk += 32) {
      {
        int row = t >> 2, c0 = (t & 3) * 8;
        const _Float16* s1 = Arm + (size_t)(rowBase + row) * MN + m * NH + kk + c0;
        const _Float16* s2 = Aim + (size_t)(rowBase + row) * MN + m * NH + kk + c0;
#pragma unroll
        for (int q = 0; q < 8; ++q) { Asr[row][c0 + q] = s1[q]; Asi[row][c0 + q] = s2[q]; }
      }
      {
        int o = t & 63, kr = t >> 6;
#pragma unroll
        for (int q = 0; q < 8; ++q) {
          int kv = kr * 8 + q;
          Bsr[o][kv] = Br[(size_t)(kk + kv) * NH + oBase + o];
          Bsi[o][kv] = Bi[(size_t)(kk + kv) * NH + oBase + o];
        }
      }
      __syncthreads();
      v16h ar = load_frag_a(&Asr[wm * 16][0], 40);
      v16h ai = load_frag_a(&Asi[wm * 16][0], 40);
#pragma unroll
      for (int nt = 0; nt < 2; ++nt) {
        v16h br = load_frag_b(&Bsr[wn * 32 + nt * 16][0], 40);
        v16h bi = load_frag_b(&Bsi[wn * 32 + nt * 16][0], 40);
        aRR[nt] = wmma_f16(ar, br, aRR[nt]);
        aII[nt] = wmma_f16(ai, bi, aII[nt]);
        aRI[nt] = wmma_f16(ar, bi, aRI[nt]);
        aIR[nt] = wmma_f16(ai, br, aIR[nt]);
      }
      __syncthreads();
    }
    float theta = 6.283185307179586f * (float)m * (float)(LSEQ - 1) / (float)LSEQ;
    float base = (m == 0 ? 1.0f : 2.0f) / (float)LSEQ;
    float cC = base * cosf(theta), cS = base * sinf(theta);
#pragma unroll
    for (int nt = 0; nt < 2; ++nt) {
#pragma unroll
      for (int i = 0; i < 8; ++i) {
        float re = aRR[nt][i] - aII[nt][i];
        float im = aRI[nt][i] + aIR[nt][i];
        int row = rowBase + wm * 16 + i + 8 * h;
        int col = oBase + wn * 32 + nt * 16 + nn;
        atomicAdd(&deck[(size_t)row * NH + col], cC * re - cS * im);
      }
    }
  }
}

// ---------------------------------------------------------------------------
// 6) eval/MLP GEMM with fused denorm + exact GELU:
//    FIN = DECh (1024x384) @ EWh (384x512) + mlp_b ; write Gm[b][s*CIN+c]
//    128x64 block, 4 WMMA per A-fragment per chunk.
// ---------------------------------------------------------------------------
__global__ __launch_bounds__(256) void gemm_eval_kernel(
    const _Float16* __restrict__ DECh, const _Float16* __restrict__ EWh,
    const float* __restrict__ mlp_b, const float* __restrict__ aw,
    const float* __restrict__ ab, const float* __restrict__ meanb,
    const float* __restrict__ stdb, float* __restrict__ Gm) {
  int sBase = blockIdx.x * 64;
  int rowBase = blockIdx.y * 128;

  __shared__ _Float16 As[128][40];
  __shared__ _Float16 Bs[64][40];

  int t = threadIdx.x, wave = t >> 5, lane = t & 31;

  v8f acc[4];
#pragma unroll
  for (int q = 0; q < 4; ++q) acc[q] = zero8();

  int rowA = t >> 1, cA = (t & 1) * 16;
  int nB = t & 63, krB = t >> 6;

  for (int kk = 0; kk < KN; kk += 32) {
    const _Float16* srcA = DECh + (size_t)(rowBase + rowA) * KN + kk + cA;
#pragma unroll
    for (int q = 0; q < 16; ++q) As[rowA][cA + q] = srcA[q];
#pragma unroll
    for (int q = 0; q < 8; ++q) {
      int kv = krB * 8 + q;
      Bs[nB][kv] = EWh[(size_t)(kk + kv) * LSEQ + sBase + nB];
    }
    if (kk + 32 < KN) {
      __builtin_prefetch(srcA + 32, 0, 0);
      __builtin_prefetch(&EWh[(size_t)(kk + 32 + krB * 8) * LSEQ + sBase + nB], 0, 0);
    }
    __syncthreads();
    v16h aF = load_frag_a(&As[wave * 16][0], 40);
#pragma unroll
    for (int nt = 0; nt < 4; ++nt) {
      v16h bF = load_frag_b(&Bs[nt * 16][0], 40);
      acc[nt] = wmma_f16(aF, bF, acc[nt]);
    }
    __syncthreads();
  }
  int h = lane >> 4, nn = lane & 15;
  float bb = mlp_b[0];
#pragma unroll
  for (int i = 0; i < 8; ++i) {
    int row = rowBase + wave * 16 + i + 8 * h;
    int b = row >> 5, c = row & 31;
    float inv = 1.0f / (aw[c] + 1e-10f);
    float sd = stdb[row], mu = meanb[row];
    float abb = ab[c];
#pragma unroll
    for (int nt = 0; nt < 4; ++nt) {
      float v = acc[nt][i] + bb;
      int s = sBase + nt * 16 + nn;
      float xg = (v - abb) * inv * sd + mu;
      float g = 0.5f * xg * (1.0f + erff(xg * 0.70710678118654752f));
      Gm[(size_t)b * FINF + s * CIN + c] = g;
    }
  }
}

// ---------------------------------------------------------------------------
// 7) final projection: logits[b][j] = Gm[b,:] . proj_w[j,:] + proj_b[j]
// ---------------------------------------------------------------------------
__global__ __launch_bounds__(256) void proj_kernel(
    const float* __restrict__ Gm, const float* __restrict__ pw,
    const float* __restrict__ pb, float* __restrict__ out) {
  int b = blockIdx.x, j = blockIdx.y;
  const float* g = Gm + (size_t)b * FINF;
  const float* w = pw + (size_t)j * FINF;
  float s = 0.f;
  for (int f = threadIdx.x; f < FINF; f += 256) s += g[f] * w[f];
  __shared__ float red[256];
  red[threadIdx.x] = s; __syncthreads();
  for (int o = 128; o > 0; o >>= 1) {
    if (threadIdx.x < o) red[threadIdx.x] += red[threadIdx.x + o];
    __syncthreads();
  }
  if (threadIdx.x == 0) out[b * NCLS + j] = red[0] + pb[j];
}

// ---------------------------------------------------------------------------
extern "C" void kernel_launch(void* const* d_in, const int* in_sizes, int n_in,
                              void* d_out, int out_size, void* d_ws, size_t ws_size,
                              hipStream_t stream) {
  const float* x_enc   = (const float*)d_in[0];
  const float* aw      = (const float*)d_in[1];
  const float* ab      = (const float*)d_in[2];
  const float* A_stack = (const float*)d_in[3];
  const float* B_stack = (const float*)d_in[4];
  const float* eval_l  = (const float*)d_in[5];
  const float* spec_wr = (const float*)d_in[6];
  const float* spec_wi = (const float*)d_in[7];
  const float* mlp_w   = (const float*)d_in[8];
  const float* mlp_b   = (const float*)d_in[9];
  const float* proj_w  = (const float*)d_in[10];
  const float* proj_b  = (const float*)d_in[11];
  float* out = (float*)d_out;

  // workspace carve-up (~87 MB total; all intermediates L2-resident)
  char* base = (char*)d_ws;
  size_t off = 0;
  auto alloc = [&](size_t bytes) -> void* {
    void* p = base + off;
    off = (off + bytes + 255) & ~(size_t)255;
    return p;
  };
  _Float16* xnh  = (_Float16*)alloc((size_t)RROWS * LSEQ * 2);
  float*    meanb = (float*)alloc((size_t)RROWS * 4);
  float*    stdb  = (float*)alloc((size_t)RROWS * 4);
  float*    Gbuf  = (float*)alloc((size_t)KS * LSEQ * NH * 4);
  _Float16* T1r  = (_Float16*)alloc((size_t)KS * LSEQ * MN * 2);
  _Float16* T1i  = (_Float16*)alloc((size_t)KS * LSEQ * MN * 2);
  _Float16* XFr  = (_Float16*)alloc((size_t)KS * RROWS * MN * 2);
  _Float16* XFi  = (_Float16*)alloc((size_t)KS * RROWS * MN * 2);
  _Float16* Whr  = (_Float16*)alloc((size_t)KS * MODES * NH * NH * 2);
  _Float16* Whi  = (_Float16*)alloc((size_t)KS * MODES * NH * NH * 2);
  float*    decbuf = (float*)alloc((size_t)KS * RROWS * NH * 4);
  _Float16* DECh = (_Float16*)alloc((size_t)RROWS * KN * 2);
  _Float16* EWh  = (_Float16*)alloc((size_t)KN * LSEQ * 2);
  float*    Gm   = (float*)alloc((size_t)BATCHN * FINF * 4);

  zero_kernel<<<(KS * RROWS * NH + 255) / 256, 256, 0, stream>>>(
      decbuf, KS * RROWS * NH);
  norm_kernel<<<RROWS, 256, 0, stream>>>(x_enc, aw, ab, xnh, meanb, stdb);
  hippo_g_kernel<<<KS, 512, 0, stream>>>(A_stack, B_stack, Gbuf);
  t1_kernel<<<dim3(16, KS), 256, 0, stream>>>(Gbuf, T1r, T1i);
  wconv_kernel<<<(int)(((size_t)KS * MODES * NH * NH + 255) / 256), 256, 0,
                 stream>>>(spec_wr, spec_wi, Whr, Whi);
  ew_kernel<<<(KN * LSEQ + 255) / 256, 256, 0, stream>>>(eval_l, mlp_w, EWh);
  gemm_dft_kernel<<<dim3(MN / 64, RROWS / 128, 2 * KS), 256, 0, stream>>>(
      xnh, T1r, T1i, XFr, XFi);
  gemm_spec_kernel<<<dim3(32, MODES / 4, KS), 256, 0, stream>>>(
      XFr, XFi, Whr, Whi, decbuf);
  dec2h_kernel<<<(RROWS * KN + 255) / 256, 256, 0, stream>>>(decbuf, DECh);
  gemm_eval_kernel<<<dim3(LSEQ / 64, RROWS / 128), 256, 0, stream>>>(
      DECh, EWh, mlp_b, aw, ab, meanb, stdb, Gm);
  proj_kernel<<<dim3(BATCHN, NCLS), 256, 0, stream>>>(Gm, proj_w, proj_b, out);
}
